// MinkEngResBlock_68676527063293
// MI455X (gfx1250) — matrix operand order repaired and checked by reading, not compiled, not run on hardware
//
#include <hip/hip_runtime.h>
#include <stdint.h>

typedef __attribute__((ext_vector_type(16))) __bf16 v16bf;
typedef __attribute__((ext_vector_type(8)))  __bf16 v8bf;
typedef __attribute__((ext_vector_type(8)))  float  v8f;
typedef __attribute__((ext_vector_type(4)))  unsigned int v4u;
typedef __attribute__((ext_vector_type(8)))  int    v8i;
typedef __attribute__((ext_vector_type(4)))  int    v4i;

#define NVOX 131072
#define KNBR 27
#define CH   64
#define ROWS_PER_BLOCK 128
#define LDS_ROW 72                   // 64 data + 8 pad bf16 (144 B) per weight row
#define SLICE_ELEMS (CH * LDS_ROW)   // 4608 bf16 = 9216 B per k-slice

#if __has_builtin(__builtin_amdgcn_tensor_load_to_lds) && __has_builtin(__builtin_amdgcn_s_wait_tensorcnt)
#define USE_TDM 1
#else
#define USE_TDM 0
#endif

// ---------------- TDM: DMA one 64x64 bf16 weight slice into LDS ------------
// 1-D tile of 4096 x 2B elements; LDS padding: +4 dwords after every 32 dwords
// (so each 128B weight row lands at a 144B stride -> 16B-aligned, bank-spread).
__device__ __forceinline__ void tdm_load_slice(const __bf16* gsrc, __bf16* ldst) {
#if USE_TDM
  unsigned lds_addr = (unsigned)(uintptr_t)ldst;           // low 32 bits = LDS byte offset
  unsigned long long ga = (unsigned long long)(uintptr_t)gsrc;
  v4u g0;
  g0.x = 1u;                                               // count=1, user mode
  g0.y = lds_addr;                                         // lds_addr [63:32]
  g0.z = (unsigned)ga;                                     // global_addr [95:64]
  g0.w = (unsigned)((ga >> 32) & 0x01FFFFFFull) | (2u << 30); // addr[56:32] | type=2
  v8i g1;
  // data_size=2B | pad_enable | pad_interval=32dw (code 4) | pad_amount=4dw (code 3)
  g1[0] = (int)((1u << 16) | (1u << 20) | (4u << 22) | (3u << 25));
  g1[1] = (int)(((unsigned)(CH * CH) & 0xFFFFu) << 16);    // tensor_dim0 = 4096 (lo16)
  g1[2] = (int)(1u << 16);                                 // tensor_dim0 hi=0, tensor_dim1=1
  g1[3] = (int)(((unsigned)(CH * CH)) << 16);              // tile_dim0 = 4096
  g1[4] = 1;                                               // tile_dim1 = 1, tile_dim2 = 0
  g1[5] = CH * CH;                                         // tensor_dim0_stride = 4096
  g1[6] = 0;
  g1[7] = 0;
  v4i gz4 = {0, 0, 0, 0};
  v8i gz8 = {0, 0, 0, 0, 0, 0, 0, 0};
  // 6-arg variant (this toolchain): (g0, g1, g2, g3, g4, cpol)
  __builtin_amdgcn_tensor_load_to_lds(g0, g1, gz4, gz4, gz8, 0);
#else
  (void)gsrc; (void)ldst;
#endif
}

__device__ __forceinline__ void tdm_wait(int pending) {
#if USE_TDM
  if (pending) __builtin_amdgcn_s_wait_tensorcnt(1);
  else         __builtin_amdgcn_s_wait_tensorcnt(0);
#endif
}

// ---------------- WMMA helpers --------------------------------------------
__device__ __forceinline__ v8f wmma_bf16(v16bf a, v16bf b, v8f c) {
  // (neg_a, A, neg_b, B, c_mod, C, reuse_a, reuse_b)
  return __builtin_amdgcn_wmma_f32_16x16x32_bf16(false, a, false, b, (short)0, c,
                                                 false, false);
}

union AFrag {
  v16bf v;
  struct { v8bf lo, hi; } h;
};

// Per-lane A fragment (ISA 16-bit A 16x32 layout):
// lane<16: row=lane,   VGPR0-3 = K 0..7,  VGPR4-7 = K 16..23 (h8 = 0)
// lane>=16: row=lane-16, VGPR0-3 = K 8..15, VGPR4-7 = K 24..31 (h8 = 8)
__device__ __forceinline__ void load_afrag(const __bf16* base, int h8,
                                           AFrag& a0, AFrag& a1) {
  a0.h.lo = *(const v8bf*)(base + h8);           // channels [h8, h8+8)
  a0.h.hi = *(const v8bf*)(base + 16 + h8);      // channels [16+h8, ...)
  a1.h.lo = *(const v8bf*)(base + 32 + h8);      // second K32 chunk
  a1.h.hi = *(const v8bf*)(base + 48 + h8);
}

__device__ __forceinline__ v16bf load_bfrag(const __bf16* p) {
  v16bf t;
  #pragma unroll
  for (int s = 0; s < 8; ++s) {
    t[s]     = p[s * LDS_ROW];         // K = h8 + s
    t[s + 8] = p[(16 + s) * LDS_ROW];  // K = 16 + h8 + s
  }
  return t;
}

// ---------------- fp32 -> bf16 convert ------------------------------------
__global__ void f32_to_bf16_kernel(const float* __restrict__ s,
                                   __bf16* __restrict__ d, int n) {
  int i = (blockIdx.x * blockDim.x + threadIdx.x) * 4;
  if (i + 3 < n) {
    float4 f = *(const float4*)(s + i);
    d[i + 0] = (__bf16)f.x;
    d[i + 1] = (__bf16)f.y;
    d[i + 2] = (__bf16)f.z;
    d[i + 3] = (__bf16)f.w;
  } else {
    for (; i < n; ++i) d[i] = (__bf16)s[i];
  }
}

// ---------------- gather-GEMM sparse conv ---------------------------------
// MODE 0: y = relu(conv(X) + bias)            -> bf16 out (hidden layer)
// MODE 1: y = relu(conv(X) + bias + resid)    -> fp32 out (residual block out)
template <int MODE>
__global__ __launch_bounds__(256) void sparse_conv_wmma(
    const __bf16* __restrict__ X,      // [N, 64] bf16 features
    const __bf16* __restrict__ Wg,     // [27, 64, 64] bf16 weights (k, cin, cout)
    const float*  __restrict__ bias,   // [64]
    const int*    __restrict__ nbr,    // [N, 27]
    __bf16* __restrict__ Ybf,          // MODE 0 output
    const float* __restrict__ resid,   // MODE 1 residual input
    float* __restrict__ Yf)            // MODE 1 output
{
  __shared__ __bf16 sW[2][SLICE_ELEMS];

  const int tid   = threadIdx.x;
  const int lane  = tid & 31;
  const int wid   = tid >> 5;
  const int mbase = blockIdx.x * ROWS_PER_BLOCK + wid * 16;  // wave's 16-row tile
  const int arow  = mbase + (lane & 15);                     // gather row for A frags
  const int h8    = (lane >> 4) << 3;                        // A-layout half select

  v8f acc0 = {0,0,0,0,0,0,0,0};
  v8f acc1 = acc0, acc2 = acc0, acc3 = acc0;

  if (USE_TDM && wid == 0) tdm_load_slice(Wg, &sW[0][0]);    // prologue: stage k=0

  AFrag a0, a1, a0n, a1n;
  {
    int idx = nbr[arow * KNBR];
    load_afrag(X + (size_t)idx * CH, h8, a0n, a1n);
  }

  for (int k = 0; k < KNBR; ++k) {
    a0 = a0n; a1 = a1n;
    const __bf16* wk;

    if (USE_TDM) {
      const int has_next = (k + 1 < KNBR);
      if (has_next && wid == 0)
        tdm_load_slice(Wg + (size_t)(k + 1) * CH * CH, &sW[(k + 1) & 1][0]);
      if (has_next) {
        int idx = nbr[arow * KNBR + k + 1];
        load_afrag(X + (size_t)idx * CH, h8, a0n, a1n);
      }
      if (wid == 0) tdm_wait(has_next);   // current slice landed (TDM in-order)
      __syncthreads();                    // publish LDS to all waves
      wk = &sW[k & 1][0];
    } else {
      __syncthreads();                    // previous readers done
      const __bf16* g = Wg + (size_t)k * CH * CH;
      #pragma unroll
      for (int i = 0; i < 2; ++i) {       // 512 x 16B chunks, 2 per thread
        int c = tid + i * 256;
        int r = c >> 3, col = (c & 7) << 3;
        *(v8bf*)(&sW[0][r * LDS_ROW + col]) = *(const v8bf*)(g + r * CH + col);
      }
      if (k + 1 < KNBR) {
        int idx = nbr[arow * KNBR + k + 1];
        load_afrag(X + (size_t)idx * CH, h8, a0n, a1n);
      }
      __syncthreads();
      wk = &sW[0][0];
    }

    // B fragments: lane holds column n = nt*16 + lane%16, elements over K
    // in table order (K rows h8.., 16+h8..; +32 for cc=1).
    const __bf16* r0 = wk + h8 * LDS_ROW + (lane & 15);
    const __bf16* r1 = wk + (32 + h8) * LDS_ROW + (lane & 15);
    v16bf b;
    b = load_bfrag(r0 +  0); acc0 = wmma_bf16(a0.v, b, acc0);
    b = load_bfrag(r1 +  0); acc0 = wmma_bf16(a1.v, b, acc0);
    b = load_bfrag(r0 + 16); acc1 = wmma_bf16(a0.v, b, acc1);
    b = load_bfrag(r1 + 16); acc1 = wmma_bf16(a1.v, b, acc1);
    b = load_bfrag(r0 + 32); acc2 = wmma_bf16(a0.v, b, acc2);
    b = load_bfrag(r1 + 32); acc2 = wmma_bf16(a1.v, b, acc2);
    b = load_bfrag(r0 + 48); acc3 = wmma_bf16(a0.v, b, acc3);
    b = load_bfrag(r1 + 48); acc3 = wmma_bf16(a1.v, b, acc3);
    __syncthreads();                      // buffer reuse fence
  }

  // C/D layout: VGPR j -> row (j + 8*(lane>=16)), lane%16 -> column
  const int ncol  = lane & 15;
  const int rbase = mbase + h8;
  const float bv0 = bias[ncol];
  const float bv1 = bias[16 + ncol];
  const float bv2 = bias[32 + ncol];
  const float bv3 = bias[48 + ncol];

  #pragma unroll
  for (int j = 0; j < 8; ++j) {
    size_t r = (size_t)(rbase + j) * CH;
    float v0 = acc0[j] + bv0;
    float v1 = acc1[j] + bv1;
    float v2 = acc2[j] + bv2;
    float v3 = acc3[j] + bv3;
    if (MODE == 0) {
      Ybf[r + ncol]      = (__bf16)fmaxf(v0, 0.0f);
      Ybf[r + 16 + ncol] = (__bf16)fmaxf(v1, 0.0f);
      Ybf[r + 32 + ncol] = (__bf16)fmaxf(v2, 0.0f);
      Ybf[r + 48 + ncol] = (__bf16)fmaxf(v3, 0.0f);
    } else {
      Yf[r + ncol]      = fmaxf(v0 + resid[r + ncol],      0.0f);
      Yf[r + 16 + ncol] = fmaxf(v1 + resid[r + 16 + ncol], 0.0f);
      Yf[r + 32 + ncol] = fmaxf(v2 + resid[r + 32 + ncol], 0.0f);
      Yf[r + 48 + ncol] = fmaxf(v3 + resid[r + 48 + ncol], 0.0f);
    }
  }
}

// ---------------- launch ---------------------------------------------------
extern "C" void kernel_launch(void* const* d_in, const int* in_sizes, int n_in,
                              void* d_out, int out_size, void* d_ws, size_t ws_size,
                              hipStream_t stream) {
  const float* feat = (const float*)d_in[0];
  const float* W1   = (const float*)d_in[1];
  const float* b1   = (const float*)d_in[2];
  const float* W2   = (const float*)d_in[3];
  const float* b2   = (const float*)d_in[4];
  const int*   nbr  = (const int*)d_in[5];

  // workspace layout (all 16B aligned)
  __bf16* feat16 = (__bf16*)d_ws;                       // N*64
  __bf16* h16    = feat16 + (size_t)NVOX * CH;          // N*64
  __bf16* w116   = h16 + (size_t)NVOX * CH;             // 27*64*64
  __bf16* w216   = w116 + KNBR * CH * CH;               // 27*64*64

  const int nf = NVOX * CH;
  const int nw = KNBR * CH * CH;
  f32_to_bf16_kernel<<<(nf / 4 + 255) / 256, 256, 0, stream>>>(feat, feat16, nf);
  f32_to_bf16_kernel<<<(nw / 4 + 255) / 256, 256, 0, stream>>>(W1, w116, nw);
  f32_to_bf16_kernel<<<(nw / 4 + 255) / 256, 256, 0, stream>>>(W2, w216, nw);

  const int grid = NVOX / ROWS_PER_BLOCK;   // 1024
  // h = relu(conv1(feat) + b1)   -> bf16
  sparse_conv_wmma<0><<<grid, 256, 0, stream>>>(feat16, w116, b1, nbr,
                                                h16, nullptr, nullptr);
  // out = relu(conv2(h) + b2 + feat) -> fp32
  sparse_conv_wmma<1><<<grid, 256, 0, stream>>>(h16, w216, b2, nbr,
                                                nullptr, feat, (float*)d_out);
}
